// DynamicMoELayer_36481452213065
// MI455X (gfx1250) — matrix operand order repaired and checked by reading, not compiled, not run on hardware
//
#include <hip/hip_runtime.h>
#include <math.h>

// Problem constants (match reference)
#define B_ 4
#define S_ 4096
#define H_ 1024
#define I_ 4096
#define E_ 8
#define T_ (B_ * S_)          // 16384 tokens
#define MT 32                 // tokens per FFN block (2 x 16-row WMMA subtiles)
#define NWAVES 16
#define BLOCK (NWAVES * 32)   // 512 threads, wave32

typedef __attribute__((ext_vector_type(16))) __bf16 v16bf;
typedef __attribute__((ext_vector_type(8)))  __bf16 v8bf;
typedef __attribute__((ext_vector_type(8)))  float  v8f;
typedef __attribute__((ext_vector_type(16))) float  v16f;

union BF16Frag { v16bf v; v8bf h[2]; };

// -------------------------------------------------------------------------
// Router: logits = x @ Wg^T ; softmax -> top2 -> renormalized gates;
// builds per-expert gathered token lists (atomic counters).
// -------------------------------------------------------------------------
__global__ __launch_bounds__(256)
void router_kernel(const float* __restrict__ x, const float* __restrict__ Wg,
                   float* __restrict__ logits, float* __restrict__ gates,
                   int* __restrict__ counts, int* __restrict__ lists)
{
  __shared__ float wg_s[E_ * H_];   // 32 KB
  const int tid = threadIdx.x;
  for (int i = tid; i < E_ * H_; i += 256) wg_s[i] = Wg[i];
  __syncthreads();

  const int lane = tid & 31, wave = tid >> 5;
  for (int j = 0; j < 8; ++j) {
    const int t = blockIdx.x * 64 + wave * 8 + j;
    const float* xr = x + (size_t)t * H_;
    float acc[E_];
#pragma unroll
    for (int e = 0; e < E_; ++e) acc[e] = 0.f;
    for (int h = lane; h < H_; h += 32) {
      const float xv = xr[h];
#pragma unroll
      for (int e = 0; e < E_; ++e) acc[e] += xv * wg_s[e * H_ + h];
    }
#pragma unroll
    for (int e = 0; e < E_; ++e)
#pragma unroll
      for (int off = 16; off > 0; off >>= 1)
        acc[e] += __shfl_xor(acc[e], off, 32);

    if (lane == 0) {
      float m = acc[0];
#pragma unroll
      for (int e = 1; e < E_; ++e) m = fmaxf(m, acc[e]);
      float p[E_];
#pragma unroll
      for (int e = 0; e < E_; ++e) p[e] = __expf(acc[e] - m);
      int i0 = 0;
#pragma unroll
      for (int e = 1; e < E_; ++e) if (p[e] > p[i0]) i0 = e;
      int i1 = (i0 == 0) ? 1 : 0;
#pragma unroll
      for (int e = 0; e < E_; ++e) if (e != i0 && p[e] > p[i1]) i1 = e;
      const float den = p[i0] + p[i1];
      const float g0 = p[i0] / den, g1 = p[i1] / den;
#pragma unroll
      for (int e = 0; e < E_; ++e) {
        logits[(size_t)t * E_ + e] = acc[e];
        gates[(size_t)t * E_ + e] = 0.f;
      }
      gates[(size_t)t * E_ + i0] = g0;
      gates[(size_t)t * E_ + i1] = g1;
      int s0 = atomicAdd(&counts[i0], 1); lists[i0 * T_ + s0] = t;
      int s1 = atomicAdd(&counts[i1], 1); lists[i1 * T_ + s1] = t;
    }
  }
}

// -------------------------------------------------------------------------
// Bulk f32 -> bf16 pre-conversion (8 elems/thread, packed v_cvt_pk_bf16_f32)
// -------------------------------------------------------------------------
__global__ __launch_bounds__(256)
void cvt_bf16_kernel(const float* __restrict__ src, __bf16* __restrict__ dst, long n8)
{
  const long i = (long)blockIdx.x * blockDim.x + threadIdx.x;
  if (i >= n8) return;
  const float4* p = (const float4*)(src + i * 8);
  float4 a = p[0], b = p[1];
  v8f t = { a.x, a.y, a.z, a.w, b.x, b.y, b.z, b.w };
  *(v8bf*)(dst + i * 8) = __builtin_convertvector(t, v8bf);
}

// -------------------------------------------------------------------------
// Fused expert FFN over gathered 32-token tiles, bf16 WMMA.
// hm (32 x 4096 bf16 = 256 KB) lives in LDS, XOR-swizzled per 8-elem chunk.
// Templated on element type of x/W1/W2: __bf16 (pre-converted, no VALU cvt
// in the hot loop) or float (convert-on-the-fly fallback).
// -------------------------------------------------------------------------
__device__ __forceinline__ int hm_off(int row, int k) {
  return row * I_ + ((((k >> 3) ^ (row & 15)) << 3) | (k & 7));
}

// A-fragment (16x32 bf16): per lane K = kb+{0..7} U {16+kb..23+kb}
__device__ __forceinline__ v16bf load_a_any(const float* __restrict__ rp) {
  const float4* p1 = (const float4*)(rp);
  const float4* p2 = (const float4*)(rp + 16);
  float4 q0 = p1[0], q1 = p1[1], q2 = p2[0], q3 = p2[1];
  v16f t = { q0.x, q0.y, q0.z, q0.w, q1.x, q1.y, q1.z, q1.w,
             q2.x, q2.y, q2.z, q2.w, q3.x, q3.y, q3.z, q3.w };
  return __builtin_convertvector(t, v16bf);
}
__device__ __forceinline__ v16bf load_a_any(const __bf16* __restrict__ rp) {
  BF16Frag f;
  f.h[0] = *(const v8bf*)(rp);
  f.h[1] = *(const v8bf*)(rp + 16);
  return f.v;
}

// B-fragment (32x16 bf16): per lane K = kbB+{0..7} U {16+kbB..23+kbB}
__device__ __forceinline__ v16bf load_b_any(const float* __restrict__ rp) {
  const float4* p = (const float4*)(rp);
  const float4* p2 = (const float4*)(rp + 16);
  float4 q0 = p[0], q1 = p[1], q2 = p2[0], q3 = p2[1];
  v16f t = { q0.x, q0.y, q0.z, q0.w, q1.x, q1.y, q1.z, q1.w,
             q2.x, q2.y, q2.z, q2.w, q3.x, q3.y, q3.z, q3.w };
  return __builtin_convertvector(t, v16bf);
}
__device__ __forceinline__ v16bf load_b_any(const __bf16* __restrict__ rp) {
  BF16Frag f;
  f.h[0] = *(const v8bf*)(rp);
  f.h[1] = *(const v8bf*)(rp + 16);
  return f.v;
}

template <typename ET>
__global__ __launch_bounds__(BLOCK, 1)
void moe_ffn_kernel(const ET* __restrict__ x,
                    const ET* __restrict__ W1, const float* __restrict__ b1,
                    const ET* __restrict__ W2, const float* __restrict__ b2,
                    const float* __restrict__ gates,
                    const int* __restrict__ counts,
                    const int* __restrict__ lists,
                    float* __restrict__ out)
{
  extern __shared__ __bf16 hm[];     // MT x I_ bf16, swizzled (256 KB)
  __shared__ int   tok_s[MT];
  __shared__ float gate_s[MT];

  const int e     = blockIdx.y;
  const int count = counts[e];
  const int tile0 = blockIdx.x * MT;
  if (tile0 >= count) return;

  const int tid  = threadIdx.x;
  const int lane = tid & 31;
  const int wave = tid >> 5;

  if (tid < MT) {
    const int idx = tile0 + tid;
    const int t = lists[e * T_ + (idx < count ? idx : tile0)];
    tok_s[tid]  = t;
    gate_s[tid] = (idx < count) ? gates[(size_t)t * E_ + e] : 0.f;
  }
  __syncthreads();

  const int ln16 = lane & 15;
  const int hi   = (lane >> 4) & 1;
  const int kbA  = hi * 8;    // A-frag K base for this half-wave
  const int kbB  = hi * 8;    // B-frag K base

  const ET*    W1e = W1 + (size_t)e * I_ * H_;
  const float* b1e = b1 + (size_t)e * I_;
  const ET*    xr0 = x + (size_t)tok_s[ln16] * H_;
  const ET*    xr1 = x + (size_t)tok_s[ln16 + 16] * H_;

  // ---------------- GEMM1 + SiLU: hm = silu(x_tile @ W1e^T + b1e) ----------
  for (int g = 0; g < 4; ++g) {
    const int nt0 = wave * 16 + g * 4;    // 4 I-tiles of 16 per group
    v8f acc[4][2] = {};
    const ET* w1r[4];
#pragma unroll
    for (int j = 0; j < 4; ++j)
      w1r[j] = W1e + (size_t)((nt0 + j) * 16 + ln16) * H_;

    // No k-unroll: body already has 8 WMMA + 12 b128 loads; unrolling spills.
#pragma clang loop unroll(disable)
    for (int k0 = 0; k0 < H_; k0 += 32) {
      v16bf a0 = load_a_any(xr0 + k0 + kbA);
      v16bf a1 = load_a_any(xr1 + k0 + kbA);
#pragma unroll
      for (int j = 0; j < 4; ++j) {
        v16bf b = load_b_any(w1r[j] + k0 + kbB);
        acc[j][0] = __builtin_amdgcn_wmma_f32_16x16x32_bf16(false, a0, false, b, (short)0, acc[j][0], false, false);
        acc[j][1] = __builtin_amdgcn_wmma_f32_16x16x32_bf16(false, a1, false, b, (short)0, acc[j][1], false, false);
      }
    }
#pragma unroll
    for (int j = 0; j < 4; ++j) {
      const int col = (nt0 + j) * 16 + ln16;
      const float bias = b1e[col];
#pragma unroll
      for (int msub = 0; msub < 2; ++msub) {
#pragma unroll
        for (int r = 0; r < 8; ++r) {
          const int row = msub * 16 + hi * 8 + r;
          const float v = acc[j][msub][r] + bias;
          const float s = v / (1.f + __expf(-v));     // silu
          hm[hm_off(row, col)] = (__bf16)s;
        }
      }
    }
  }
  __syncthreads();

  // ---------------- GEMM2: out += gate * (hm @ W2e^T + b2e) ----------------
  const ET*    W2e = W2 + (size_t)e * H_ * I_;
  const float* b2e = b2 + (size_t)e * H_;
  v8f acc2[4][2] = {};
  const ET* w2r[4];
#pragma unroll
  for (int j = 0; j < 4; ++j)
    w2r[j] = W2e + (size_t)(wave * 64 + j * 16 + ln16) * I_;

#pragma clang loop unroll(disable)
  for (int k0 = 0; k0 < I_; k0 += 32) {
    BF16Frag fa0, fa1;
    fa0.h[0] = *(const v8bf*)(hm + hm_off(ln16, k0 + kbA));
    fa0.h[1] = *(const v8bf*)(hm + hm_off(ln16, k0 + kbA + 16));
    fa1.h[0] = *(const v8bf*)(hm + hm_off(ln16 + 16, k0 + kbA));
    fa1.h[1] = *(const v8bf*)(hm + hm_off(ln16 + 16, k0 + kbA + 16));
#pragma unroll
    for (int j = 0; j < 4; ++j) {
      __builtin_prefetch(w2r[j] + k0 + 256, 0, 0);  // global_prefetch_b8
      v16bf b = load_b_any(w2r[j] + k0 + kbB);
      acc2[j][0] = __builtin_amdgcn_wmma_f32_16x16x32_bf16(false, fa0.v, false, b, (short)0, acc2[j][0], false, false);
      acc2[j][1] = __builtin_amdgcn_wmma_f32_16x16x32_bf16(false, fa1.v, false, b, (short)0, acc2[j][1], false, false);
    }
  }

#pragma unroll
  for (int j = 0; j < 4; ++j) {
    const int col = wave * 64 + j * 16 + ln16;
    const float bias = b2e[col];
#pragma unroll
    for (int m = 0; m < 2; ++m) {
#pragma unroll
      for (int r = 0; r < 8; ++r) {
        const int row = m * 16 + hi * 8 + r;
        if (tile0 + row < count) {
          const float vv = (acc2[j][m][r] + bias) * gate_s[row];
          atomicAdd(out + (size_t)tok_s[row] * H_ + col, vv);  // exactly 2 adds/elem -> deterministic
        }
      }
    }
  }
}

// -------------------------------------------------------------------------
extern "C" void kernel_launch(void* const* d_in, const int* in_sizes, int n_in,
                              void* d_out, int out_size, void* d_ws, size_t ws_size,
                              hipStream_t stream)
{
  (void)in_sizes; (void)n_in; (void)out_size;
  const float* x  = (const float*)d_in[0];
  const float* Wg = (const float*)d_in[1];
  const float* W1 = (const float*)d_in[2];
  const float* b1 = (const float*)d_in[3];
  const float* W2 = (const float*)d_in[4];
  const float* b2 = (const float*)d_in[5];

  float* out    = (float*)d_out;                 // [T, H]
  float* logits = out + (size_t)T_ * H_;         // [T, E]

  // workspace: gates [T*E] f32 | counts | lists [E*T] int | (opt) bf16 mirrors
  float* gates  = (float*)d_ws;
  int*   counts = (int*)((char*)d_ws + (size_t)T_ * E_ * sizeof(float));
  int*   lists  = counts + 64;

  const size_t nX  = (size_t)T_ * H_;        // 16.8M
  const size_t nW  = (size_t)E_ * I_ * H_;   // 33.6M each for W1, W2
  const size_t base = 2u << 20;              // 2 MB for gates/counts/lists
  const size_t need = base + 2 * (nX + 2 * nW);  // bf16 mirrors (~168 MB)

  hipMemsetAsync(out, 0, (size_t)T_ * H_ * sizeof(float), stream);
  hipMemsetAsync(counts, 0, 64 * sizeof(int), stream);

  router_kernel<<<T_ / 64, 256, 0, stream>>>(x, Wg, logits, gates, counts, lists);

  const int smem = MT * I_ * (int)sizeof(unsigned short);   // 256 KB of the 320 KB WGP LDS
  dim3 grid(T_ / MT, E_);

  if (ws_size >= need) {
    // Pre-convert x/W1/W2 to bf16 once: halves weight-side L2/HBM traffic and
    // removes all converts from the WMMA hot loop. Path choice is fixed by
    // ws_size -> deterministic across calls.
    __bf16* xb  = (__bf16*)((char*)d_ws + base);
    __bf16* w1b = xb + nX;
    __bf16* w2b = w1b + nW;
    cvt_bf16_kernel<<<(int)(nX / 8 / 256), 256, 0, stream>>>(x,  xb,  (long)(nX / 8));
    cvt_bf16_kernel<<<(int)(nW / 8 / 256), 256, 0, stream>>>(W1, w1b, (long)(nW / 8));
    cvt_bf16_kernel<<<(int)(nW / 8 / 256), 256, 0, stream>>>(W2, w2b, (long)(nW / 8));
    hipFuncSetAttribute(reinterpret_cast<const void*>(moe_ffn_kernel<__bf16>),
                        hipFuncAttributeMaxDynamicSharedMemorySize, smem);
    moe_ffn_kernel<__bf16><<<grid, BLOCK, smem, stream>>>(xb, w1b, b1, w2b, b2,
                                                          gates, counts, lists, out);
  } else {
    hipFuncSetAttribute(reinterpret_cast<const void*>(moe_ffn_kernel<float>),
                        hipFuncAttributeMaxDynamicSharedMemorySize, smem);
    moe_ffn_kernel<float><<<grid, BLOCK, smem, stream>>>(x, W1, b1, W2, b2,
                                                         gates, counts, lists, out);
  }
}
